// LocalMixtureNN_35347580846481
// MI455X (gfx1250) — compile-verified, matrix operand order checked by reading, not run on hardware
//
#include <hip/hip_runtime.h>
#include <math.h>

// ---------------------------------------------------------------------------
// LocalMixtureNN on MI455X (gfx1250). All big GEMMs via v_wmma_f32_16x16x32_f16,
// register-blocked so A fragments are amortized over 4 WMMA ops.
// ---------------------------------------------------------------------------

typedef _Float16 f16;
typedef _Float16 v16h __attribute__((ext_vector_type(16)));
typedef float    v8f  __attribute__((ext_vector_type(8)));

#define EMB_    300
#define EMBP    320      // padded K (multiple of 32)
#define AUD_    74
#define HID_    128
#define G4_     512      // 4*HID
#define DIMQ    256      // D1*D2
#define KC_     512      // 2*DIMQ (concat real|imag)
#define UNITS_  128
#define CELL_   64
#define B_      32
#define S_      32
#define NTOK    1024

struct __align__(16) U4 { unsigned int a, b, c, d; };
union FragU { v16h v; U4 q[2]; };

// Load one 16x32 f16 operand fragment (A from rows of X, or B from rows of W
// when computing X @ W^T). Lane l: row = row0 + l%16, K-chunks (l/16)*8 and +16.
__device__ __forceinline__ v16h ld_frag(const f16* base, int ld, int row0, int k0) {
    int lane = threadIdx.x & 31;
    const f16* p = base + (size_t)(row0 + (lane & 15)) * ld + (k0 + ((lane >> 4) << 3));
    FragU f;
    f.q[0] = *(const U4*)(p);
    f.q[1] = *(const U4*)(p + 16);
    return f.v;
}

__device__ __forceinline__ v8f wmma16(v16h a, v16h b, v8f c) {
    return __builtin_amdgcn_wmma_f32_16x16x32_f16(false, a, false, b, (short)0, c, false, false);
}

__device__ __forceinline__ float sigf(float x) { return 1.0f / (1.0f + __expf(-x)); }

// ---------------------------------------------------------------- prep kernels
__global__ void k_conv_wih(const float* w_ih, f16* Wih) {     // 512x300 -> 512x320 (pad 0)
    int g = blockIdx.x * blockDim.x + threadIdx.x;
    if (g >= G4_ * EMBP) return;
    int r = g / EMBP, c = g % EMBP;
    Wih[g] = (c < EMB_) ? (f16)w_ih[r * EMB_ + c] : (f16)0.0f;
}

__global__ void k_conv_whh(const float* w_hh, f16* Whh) {     // 512x128
    int g = blockIdx.x * blockDim.x + threadIdx.x;
    if (g >= G4_ * HID_) return;
    Whh[g] = (f16)w_hh[g];
}

__global__ void k_embed(const int* idx, const float* lut, f16* Xf) {  // token-major, padded
    int t = blockIdx.x, c = threadIdx.x;                      // 1024 blocks x 320 threads
    int wi = idx[t];
    Xf[t * EMBP + c] = (c < EMB_) ? (f16)lut[(size_t)wi * EMB_ + c] : (f16)0.0f;
}

// Normalize measurement vectors, build Wa=[Vr|Vi], Wb=[-Vi|Vr] (128x512 f16)
__global__ void k_meas_prep(const float* mr, const float* mi, f16* Wa, f16* Wb) {
    __shared__ float red[256];
    int u = blockIdx.x, d = threadIdx.x;
    float a = mr[u * DIMQ + d], b = mi[u * DIMQ + d];
    red[d] = a * a + b * b;
    __syncthreads();
    for (int off = 128; off > 0; off >>= 1) { if (d < off) red[d] += red[d + off]; __syncthreads(); }
    float kn = sqrtf(red[0]) + 1e-10f;
    float vr = a / kn, vi = b / kn;
    Wa[u * KC_ + d]        = (f16)vr;
    Wa[u * KC_ + DIMQ + d] = (f16)vi;
    Wb[u * KC_ + d]        = (f16)(-vi);
    Wb[u * KC_ + DIMQ + d] = (f16)vr;
}

// ------------------------------------------------- GEMM 1: G0 = X @ Wih^T + b
// M=1024 N=512 K=320. Each wave: 1 M-tile x 4 N-tiles (A fragment reused 4x).
// Grid: 64 blocks (M-tiles) x 8 waves (N-groups of 64).
__global__ __launch_bounds__(256) void k_gemm_gates(const f16* Xf, const f16* Wih,
                                                    const float* b_ih, const float* b_hh,
                                                    float* G0) {
    int wave = threadIdx.x >> 5, lane = threadIdx.x & 31;
    int m0 = blockIdx.x * 16;
    int nbase = wave * 64;
    v8f acc0 = {}, acc1 = {}, acc2 = {}, acc3 = {};
#pragma unroll
    for (int kc = 0; kc < EMBP; kc += 32) {
        v16h a  = ld_frag(Xf, EMBP, m0, kc);
        v16h b0 = ld_frag(Wih, EMBP, nbase +  0, kc);
        v16h b1 = ld_frag(Wih, EMBP, nbase + 16, kc);
        v16h b2 = ld_frag(Wih, EMBP, nbase + 32, kc);
        v16h b3 = ld_frag(Wih, EMBP, nbase + 48, kc);
        acc0 = wmma16(a, b0, acc0);
        acc1 = wmma16(a, b1, acc1);
        acc2 = wmma16(a, b2, acc2);
        acc3 = wmma16(a, b3, acc3);
    }
    int rb = m0 + ((lane >> 4) << 3);
    v8f accs[4] = {acc0, acc1, acc2, acc3};
#pragma unroll
    for (int j = 0; j < 4; ++j) {
        int col = nbase + j * 16 + (lane & 15);
        float bias = b_ih[col] + b_hh[col];
#pragma unroll
        for (int r = 0; r < 8; ++r)
            G0[(size_t)(rb + r) * G4_ + col] = accs[j][r] + bias;
    }
}

// ------------------------------------------------- GEMM 2 (recurrent): LSTM
// Single workgroup, 512 threads (16 waves). Per step: 64 WMMA tiles (32x512);
// each wave owns 1 M-tile x 4 N-tiles so the LDS A-fragment is reused 4x.
__global__ __launch_bounds__(512) void k_lstm(const float* G0, const f16* Whh, f16* Hall) {
    __shared__ __align__(16) f16 hbuf[B_ * HID_];   //  8 KB
    __shared__ float cbuf[B_ * HID_];               // 16 KB
    __shared__ float gsm[B_ * G4_];                 // 64 KB
    int tid = threadIdx.x, wave = tid >> 5, lane = tid & 31;
    int mt = wave >> 3;                  // 0..1  (M-tile)
    int nbase = (wave & 7) * 64;         // 4 N-tiles of 16

    for (int e = tid; e < B_ * HID_; e += 512) { hbuf[e] = (f16)0.0f; cbuf[e] = 0.0f; }
    __syncthreads();

    for (int s = 0; s < S_; ++s) {
        // Stage A: gates = h @ Whh^T + G0[token]
        v8f acc0 = {}, acc1 = {}, acc2 = {}, acc3 = {};
#pragma unroll
        for (int kc = 0; kc < HID_; kc += 32) {
            v16h a  = ld_frag(hbuf, HID_, mt * 16, kc);        // LDS (ds_load_b128)
            v16h b0 = ld_frag(Whh, HID_, nbase +  0, kc);      // global (L2 resident)
            v16h b1 = ld_frag(Whh, HID_, nbase + 16, kc);
            v16h b2 = ld_frag(Whh, HID_, nbase + 32, kc);
            v16h b3 = ld_frag(Whh, HID_, nbase + 48, kc);
            acc0 = wmma16(a, b0, acc0);
            acc1 = wmma16(a, b1, acc1);
            acc2 = wmma16(a, b2, acc2);
            acc3 = wmma16(a, b3, acc3);
        }
        int rb = mt * 16 + ((lane >> 4) << 3);
        v8f accs[4] = {acc0, acc1, acc2, acc3};
#pragma unroll
        for (int j = 0; j < 4; ++j) {
            int col = nbase + j * 16 + (lane & 15);
#pragma unroll
            for (int r = 0; r < 8; ++r) {
                int bb = rb + r;                               // batch row
                gsm[bb * G4_ + col] = accs[j][r] + G0[(size_t)(bb * S_ + s) * G4_ + col];
            }
        }
        __syncthreads();
        // Stage B: pointwise cell update (gate order i,f,g,o)
#pragma unroll
        for (int e = tid; e < B_ * HID_; e += 512) {
            int bb = e >> 7, j = e & 127;
            float gi = gsm[bb * G4_ + j];
            float gf = gsm[bb * G4_ + HID_ + j];
            float gg = gsm[bb * G4_ + 2 * HID_ + j];
            float go = gsm[bb * G4_ + 3 * HID_ + j];
            float c = sigf(gf) * cbuf[e] + sigf(gi) * tanhf(gg);
            float h = sigf(go) * tanhf(c);
            cbuf[e] = c;
            hbuf[e] = (f16)h;
            Hall[(size_t)(bb * S_ + s) * HID_ + j] = (f16)h;
        }
        __syncthreads();
    }
}

// --------------------------- amplitudes, phases, tensor product (per token)
__global__ __launch_bounds__(256) void k_amp_tp(
    const int* idx, const float* audio, const f16* Hall,
    const float* pt0, const float* pt1,
    const float* w_lin, const float* b_lin,
    const float* w1, const float* b1, const float* w2, const float* b2,
    const float* w3, const float* b3, const float* modw,
    f16* Xc, float* WG) {
    __shared__ float amp0[16], amp1[16], a1[16], a2[16];
    __shared__ float r0[16], i0[16], r1[16], i1[16], nn[2];
    int t = blockIdx.x, tid = threadIdx.x;
    int wi = idx[t];

    if (tid < 16) {
        float acc = b_lin[tid];
        for (int k = 0; k < HID_; ++k) acc += (float)Hall[(size_t)t * HID_ + k] * w_lin[tid * HID_ + k];
        amp0[tid] = acc;
    } else if (tid < 32) {
        int j = tid - 16;
        float acc = b1[j];
        for (int k = 0; k < AUD_; ++k) acc += audio[(size_t)t * AUD_ + k] * w1[j * AUD_ + k];
        a1[j] = fmaxf(acc, 0.0f);
    }
    __syncthreads();
    if (tid < 16) {
        float acc = b2[tid];
        for (int k = 0; k < 16; ++k) acc += a1[k] * w2[tid * 16 + k];
        a2[tid] = fmaxf(acc, 0.0f);
    }
    __syncthreads();
    if (tid < 16) {
        float acc = b3[tid];
        for (int k = 0; k < 16; ++k) acc += a2[k] * w3[tid * 16 + k];
        amp1[tid] = fmaxf(acc, 0.0f);
    }
    __syncthreads();
    if (tid == 0) {
        float s0 = 0.0f, s1 = 0.0f;
        for (int k = 0; k < 16; ++k) { s0 += amp0[k] * amp0[k]; s1 += amp1[k] * amp1[k]; }
        float n0 = sqrtf(s0), n1 = sqrtf(s1);
        nn[0] = n0; nn[1] = n1;
        float e0 = __expf(modw[0]), e1 = __expf(modw[1]);
        WG[t] = (e0 * n0 + e1 * n1) / (e0 + e1);          // softmax-mixed norm
    }
    __syncthreads();
    if (tid < 16) {
        float an = amp0[tid] / (nn[0] + 1e-10f);
        float p = pt0[(size_t)wi * 16 + tid];
        r0[tid] = an * __cosf(p); i0[tid] = an * __sinf(p);
    } else if (tid < 32) {
        int j = tid - 16;
        float an = amp1[j] / (nn[1] + 1e-10f);
        float p = pt1[(size_t)wi * 16 + j];
        r1[j] = an * __cosf(p); i1[j] = an * __sinf(p);
    }
    __syncthreads();
    // complex tensor product starting from (1 + 1i):  tp = (r0 - i0) + (r0 + i0) i
    int i = tid >> 4, j = tid & 15;
    float tpr = r0[i] - i0[i], tpi = r0[i] + i0[i];
    float re = tpr * r1[j] - tpi * i1[j];
    float im = tpr * i1[j] + tpi * r1[j];
    Xc[(size_t)t * KC_ + tid]        = (f16)re;
    Xc[(size_t)t * KC_ + DIMQ + tid] = (f16)im;
}

// --------------------- GEMM 3: alpha/beta measurement projections, S2=a^2+b^2
// M=1024 N=128 K=512. Each wave: 2 M-tiles x 1 N-tile x {alpha,beta} = 4 WMMA
// per K-chunk, sharing A fragments across the two GEMMs and B across M-tiles.
// Grid: 32 blocks (M-groups of 32 rows) x 8 waves (N-tiles).
__global__ __launch_bounds__(256) void k_meas_gemm(const f16* Xc, const f16* Wa, const f16* Wb,
                                                   float* S2) {
    int wave = threadIdx.x >> 5, lane = threadIdx.x & 31;
    int m0 = blockIdx.x * 32, n0 = wave * 16;
    v8f accA0 = {}, accA1 = {}, accB0 = {}, accB1 = {};
#pragma unroll
    for (int kc = 0; kc < KC_; kc += 32) {
        v16h a0 = ld_frag(Xc, KC_, m0,      kc);
        v16h a1 = ld_frag(Xc, KC_, m0 + 16, kc);
        v16h ba = ld_frag(Wa, KC_, n0, kc);
        v16h bb = ld_frag(Wb, KC_, n0, kc);
        accA0 = wmma16(a0, ba, accA0);
        accA1 = wmma16(a1, ba, accA1);
        accB0 = wmma16(a0, bb, accB0);
        accB1 = wmma16(a1, bb, accB1);
    }
    int col = n0 + (lane & 15);
    int rb  = m0 + ((lane >> 4) << 3);
#pragma unroll
    for (int r = 0; r < 8; ++r) {
        float al = accA0[r], be = accB0[r];
        S2[(size_t)(rb + r) * UNITS_ + col] = al * al + be * be;
        float al1 = accA1[r], be1 = accB1[r];
        S2[(size_t)(rb + 16 + r) * UNITS_ + col] = al1 * al1 + be1 * be1;
    }
}

// --------------------- n-gram softmax mixture, max-pool, final MLP head
__global__ __launch_bounds__(128) void k_final(const float* S2, const float* WG,
                                               const float* fw1, const float* fb1,
                                               const float* fw2, const float* fb2,
                                               const float* fw3, const float* fb3,
                                               float* out) {
    __shared__ float we[34], feat[UNITS_], y1[CELL_], y2[CELL_];
    int b = blockIdx.x, u = threadIdx.x;
    if (u < 34) we[u] = (u < S_) ? __expf(WG[b * S_ + u]) : 1.0f;  // pad -> exp(0)=1
    __syncthreads();
    float fm = -1e30f;
    for (int s = 0; s < S_; ++s) {
        float s2a = S2[(size_t)(b * S_ + s) * UNITS_ + u];
        fm = fmaxf(fm, s2a);                                       // 1-gram: weight==1
        float s2b = (s + 1 < S_) ? S2[(size_t)(b * S_ + s + 1) * UNITS_ + u] : 0.0f;
        float s2c = (s + 2 < S_) ? S2[(size_t)(b * S_ + s + 2) * UNITS_ + u] : 0.0f;
        float den = we[s] + we[s + 1] + we[s + 2];
        float m3 = (we[s] * s2a + we[s + 1] * s2b + we[s + 2] * s2c) / den;
        fm = fmaxf(fm, m3);
    }
    feat[u] = fm;
    __syncthreads();
    if (u < CELL_) {
        float acc = fb1[u];
        for (int k = 0; k < UNITS_; ++k) acc += feat[k] * fw1[u * UNITS_ + k];
        y1[u] = fmaxf(acc, 0.0f);
    }
    __syncthreads();
    if (u < CELL_) {
        float acc = fb2[u];
        for (int k = 0; k < CELL_; ++k) acc += y1[k] * fw2[u * CELL_ + k];
        y2[u] = fmaxf(acc, 0.0f);
    }
    __syncthreads();
    if (u == 0) {
        float acc = fb3[0];
        for (int k = 0; k < CELL_; ++k) acc += y2[k] * fw3[k];
        out[b] = acc;
    }
}

// ---------------------------------------------------------------------------
extern "C" void kernel_launch(void* const* d_in, const int* in_sizes, int n_in,
                              void* d_out, int out_size, void* d_ws, size_t ws_size,
                              hipStream_t stream) {
    (void)in_sizes; (void)n_in; (void)out_size; (void)ws_size;
    const int*   word_idx = (const int*)  d_in[0];
    const float* audio    = (const float*)d_in[1];
    const float* lut      = (const float*)d_in[2];
    const float* pt0      = (const float*)d_in[3];
    const float* pt1      = (const float*)d_in[4];
    const float* w_ih     = (const float*)d_in[5];
    const float* w_hh     = (const float*)d_in[6];
    const float* b_ih     = (const float*)d_in[7];
    const float* b_hh     = (const float*)d_in[8];
    const float* w_lin    = (const float*)d_in[9];
    const float* b_lin    = (const float*)d_in[10];
    const float* w1       = (const float*)d_in[11];
    const float* b1       = (const float*)d_in[12];
    const float* w2       = (const float*)d_in[13];
    const float* b2       = (const float*)d_in[14];
    const float* w3       = (const float*)d_in[15];
    const float* b3       = (const float*)d_in[16];
    const float* modw     = (const float*)d_in[17];
    const float* meas_r   = (const float*)d_in[18];
    const float* meas_i   = (const float*)d_in[19];
    const float* fw1      = (const float*)d_in[20];
    const float* fb1      = (const float*)d_in[21];
    const float* fw2      = (const float*)d_in[22];
    const float* fb2      = (const float*)d_in[23];
    const float* fw3      = (const float*)d_in[24];
    const float* fb3      = (const float*)d_in[25];
    float* out = (float*)d_out;

    // workspace layout (16B aligned throughout)
    char* w = (char*)d_ws;
    size_t off = 0;
    f16*   Xf   = (f16*)  (w + off); off += (size_t)NTOK * EMBP * 2;   // 640 KB
    f16*   Wih  = (f16*)  (w + off); off += (size_t)G4_ * EMBP * 2;    // 320 KB
    f16*   Whh  = (f16*)  (w + off); off += (size_t)G4_ * HID_ * 2;    // 128 KB
    float* G0   = (float*)(w + off); off += (size_t)NTOK * G4_ * 4;    //   2 MB
    f16*   Hall = (f16*)  (w + off); off += (size_t)NTOK * HID_ * 2;   // 256 KB
    f16*   Wa   = (f16*)  (w + off); off += (size_t)UNITS_ * KC_ * 2;  // 128 KB
    f16*   Wb   = (f16*)  (w + off); off += (size_t)UNITS_ * KC_ * 2;  // 128 KB
    f16*   Xc   = (f16*)  (w + off); off += (size_t)NTOK * KC_ * 2;    //   1 MB
    float* S2   = (float*)(w + off); off += (size_t)NTOK * UNITS_ * 4; // 512 KB
    float* WG   = (float*)(w + off); off += (size_t)NTOK * 4;          //   4 KB

    // prep (independent)
    k_conv_wih <<<(G4_ * EMBP + 255) / 256, 256, 0, stream>>>(w_ih, Wih);
    k_conv_whh <<<(G4_ * HID_ + 255) / 256, 256, 0, stream>>>(w_hh, Whh);
    k_embed    <<<NTOK, EMBP, 0, stream>>>(word_idx, lut, Xf);
    k_meas_prep<<<UNITS_, DIMQ, 0, stream>>>(meas_r, meas_i, Wa, Wb);

    // GEMM 1: gate inputs for all tokens
    k_gemm_gates<<<64, 256, 0, stream>>>(Xf, Wih, b_ih, b_hh, G0);

    // GEMM 2: recurrent LSTM (single WGP, state in LDS)
    k_lstm<<<1, 512, 0, stream>>>(G0, Whh, Hall);

    // pointwise: amplitudes, phases, tensor product -> [Re|Im] f16
    k_amp_tp<<<NTOK, 256, 0, stream>>>(word_idx, audio, Hall, pt0, pt1,
                                       w_lin, b_lin, w1, b1, w2, b2, w3, b3,
                                       modw, Xc, WG);

    // GEMM 3: measurement projections + |.|^2
    k_meas_gemm<<<32, 256, 0, stream>>>(Xc, Wa, Wb, S2);

    // n-gram mixture, pooling, MLP head
    k_final<<<B_, 128, 0, stream>>>(S2, WG, fw1, fb1, fw2, fb2, fw3, fb3, out);
}
